// RankingModel_v4_60722247631614
// MI455X (gfx1250) — matrix operand order repaired and checked by reading, not compiled, not run on hardware
//
#include <hip/hip_runtime.h>

typedef __attribute__((ext_vector_type(16))) _Float16 v16h;
typedef __attribute__((ext_vector_type(8)))  _Float16 v8h;
typedef __attribute__((ext_vector_type(8)))  float    v8f;

#define NB      8
#define NROWS   1536
#define NCOLS   16
#define NCLASS  32
#define NLAT    16
#define H1      32
#define H2      64
#define SORTN   2048
#define EPS_REG 0.01f
#define NTILES  (NROWS / 16)          // 96 row-tiles per batch
#define NBLOCKS (NB * NTILES)         // 768 MLP workgroups

// f16 weight-blob offsets (in _Float16 units) inside the scratch blob region.
// Layout per weight: WT[n][Kpad] (column-major vs. original), K zero-padded.
#define OFF_W1T  0      // 32 cols x Kpad32  (enc_w1 16x32)
#define OFF_W2T  1024   // 64 cols x Kpad32  (enc_w2 32x64)
#define OFF_MWT  3072   // 16 cols x Kpad64  (mean_w 64x16)
#define OFF_DW1T 4096   // 32 cols x Kpad32  (dec_w1 16x32)
#define OFF_DW2T 5120   // 16 cols x Kpad32  (dec_w2 32x16)
#define BLOB_HALFS 5632

// ---------------------------------------------------------------------------
// WMMA helper (CDNA5 gfx1250, wave32, V_WMMA_F32_16X16X32_F16)
// ---------------------------------------------------------------------------
__device__ __forceinline__ v8f wmma16(v16h a, v16h b, v8f c) {
  return __builtin_amdgcn_wmma_f32_16x16x32_f16(false, a, false, b, (short)0, c,
                                                false, false);
}

// A operand (16x32 f16, ISA 7.12.2): per-lane elements are two contiguous
// 8-half runs: k = chunk + half*8 + e (e<8) and k = chunk + 16 + half*8 + e.
// src is f16 activations in LDS, row-major with leading dim ldh (K padded).
__device__ __forceinline__ v16h loadA(const _Float16* src, int ldh, int chunk,
                                      int half, int m) {
  const _Float16* p = src + m * ldh + chunk + half * 8;
  v8h lo = *(const v8h*)p;          // ds_load_b128
  v8h hi = *(const v8h*)(p + 16);   // ds_load_b128
  return __builtin_shufflevector(lo, hi, 0, 1, 2, 3, 4, 5, 6, 7, 8, 9, 10, 11,
                                 12, 13, 14, 15);
}

// B operand (32x16 f16): lane's 16 elements k = chunk + half*8 + e (e<8) and
// k = chunk + 16 + half*8 + e for fixed column n0+m in the WT[n][Kpad] blob.
__device__ __forceinline__ v16h loadB(const _Float16* __restrict__ WT, int ldk,
                                      int n0, int chunk, int half, int m) {
  const _Float16* p = WT + (size_t)(n0 + m) * ldk + chunk + half * 8;
  v8h lo = *(const v8h*)p;
  v8h hi = *(const v8h*)(p + 16);
  return __builtin_shufflevector(lo, hi, 0, 1, 2, 3, 4, 5, 6, 7, 8, 9, 10, 11,
                                 12, 13, 14, 15);
}

// ---------------------------------------------------------------------------
// Stage 0: transpose + f16-convert + K-pad all weights into scratch (once).
// ---------------------------------------------------------------------------
__global__ __launch_bounds__(256) void prep_weights_kernel(
    const float* __restrict__ ew1, const float* __restrict__ ew2,
    const float* __restrict__ mw, const float* __restrict__ dw1,
    const float* __restrict__ dw2, _Float16* __restrict__ blob) {
  const int tid = threadIdx.x;
  // W1T[n*32+k] : enc_w1 (16x32), Kpad 32
  for (int i = tid; i < 32 * 32; i += 256) {
    int n = i >> 5, k = i & 31;
    blob[OFF_W1T + i] = (k < 16) ? (_Float16)ew1[k * 32 + n] : (_Float16)0.0f;
  }
  // W2T[n*32+k] : enc_w2 (32x64)
  for (int i = tid; i < 64 * 32; i += 256) {
    int n = i >> 5, k = i & 31;
    blob[OFF_W2T + i] = (_Float16)ew2[k * 64 + n];
  }
  // MWT[n*64+k] : mean_w (64x16)
  for (int i = tid; i < 16 * 64; i += 256) {
    int n = i >> 6, k = i & 63;
    blob[OFF_MWT + i] = (_Float16)mw[k * 16 + n];
  }
  // DW1T[n*32+k] : dec_w1 (16x32), Kpad 32
  for (int i = tid; i < 32 * 32; i += 256) {
    int n = i >> 5, k = i & 31;
    blob[OFF_DW1T + i] = (k < 16) ? (_Float16)dw1[k * 32 + n] : (_Float16)0.0f;
  }
  // DW2T[n*32+k] : dec_w2 (32x16)
  for (int i = tid; i < 16 * 32; i += 256) {
    int n = i >> 5, k = i & 31;
    blob[OFF_DW2T + i] = (_Float16)dw2[k * 16 + n];
  }
}

// ---------------------------------------------------------------------------
// Stage 1: MLP forward via WMMA. One wave (32 threads) per 16-row tile.
// ---------------------------------------------------------------------------
__global__ __launch_bounds__(32) void mlp_wmma_kernel(
    const float* __restrict__ table, const _Float16* __restrict__ blob,
    const float* __restrict__ eb1, const float* __restrict__ eb2,
    const float* __restrict__ mb, const float* __restrict__ db1,
    const float* __restrict__ db2, float* __restrict__ zbuf,
    float* __restrict__ loss_partials) {
  __shared__ __align__(16) _Float16 Xh[16][32];  // X, K padded 16->32
  __shared__ __align__(16) _Float16 Ah[16][64];  // h1(32) -> h2(64) -> r(32)
  __shared__ __align__(16) _Float16 Zh[16][32];  // z, K padded 16->32

  const int blk  = blockIdx.x;
  const int b    = blk / NTILES;
  const int r0   = (blk % NTILES) * 16;
  const int lane = threadIdx.x;
  const int half = lane >> 4;
  const int m    = lane & 15;

  const _Float16* W1T  = blob + OFF_W1T;
  const _Float16* W2T  = blob + OFF_W2T;
  const _Float16* MWT  = blob + OFF_MWT;
  const _Float16* DW1T = blob + OFF_DW1T;
  const _Float16* DW2T = blob + OFF_DW2T;

  __builtin_prefetch(blob, 0, 1);  // global_prefetch_b8 (weights -> caches)

  // Zero pad columns of Xh/Zh; load & convert the 16x16 input tile.
  const float* Xg = table + ((size_t)b * NROWS + r0) * NCOLS;
  for (int i = lane; i < 256; i += 32) {
    int r = i >> 4, c = i & 15;
    Xh[r][16 + c] = (_Float16)0.0f;
    Zh[r][16 + c] = (_Float16)0.0f;
    Xh[r][c] = (_Float16)Xg[i];
  }
  __syncthreads();

  // ---- Layer 1: h1 = relu(X @ ew1 + eb1)  (K=16 padded to 32) ----
  {
    v16h a = loadA(&Xh[0][0], 32, 0, half, m);
    v8f acc[2];
#pragma unroll
    for (int t = 0; t < 2; ++t) {
      v8f c = {};
      acc[t] = wmma16(a, loadB(W1T, 32, t * 16, 0, half, m), c);
    }
#pragma unroll
    for (int t = 0; t < 2; ++t)
#pragma unroll
      for (int r = 0; r < 8; ++r) {
        int M = r + half * 8, n = t * 16 + m;
        float v = acc[t][r] + eb1[n];
        Ah[M][n] = (_Float16)(v > 0.0f ? v : 0.0f);
      }
  }
  __syncthreads();

  // ---- Layer 2: h2 = relu(h1 @ ew2 + eb2)  (16x32 @ 32x64) ----
  {
    v16h a = loadA(&Ah[0][0], 64, 0, half, m);
    v8f acc[4];
#pragma unroll
    for (int t = 0; t < 4; ++t) {
      v8f c = {};
      acc[t] = wmma16(a, loadB(W2T, 32, t * 16, 0, half, m), c);
    }
    __syncthreads();  // finish h1 reads before overwriting Ah with h2
#pragma unroll
    for (int t = 0; t < 4; ++t)
#pragma unroll
      for (int r = 0; r < 8; ++r) {
        int M = r + half * 8, n = t * 16 + m;
        float v = acc[t][r] + eb2[n];
        Ah[M][n] = (_Float16)(v > 0.0f ? v : 0.0f);
      }
  }
  __syncthreads();

  // ---- Layer 3: z = h2 @ mean_w + mb  (K=64 -> two chained WMMAs) ----
  {
    v16h alo = loadA(&Ah[0][0], 64, 0, half, m);
    v16h ahi = loadA(&Ah[0][0], 64, 32, half, m);
    v8f c = {};
    c = wmma16(alo, loadB(MWT, 64, 0, 0, half, m), c);
    c = wmma16(ahi, loadB(MWT, 64, 0, 32, half, m), c);
    float bias = mb[m];
#pragma unroll
    for (int r = 0; r < 8; ++r) {
      int M = r + half * 8;
      float zv = c[r] + bias;
      Zh[M][m] = (_Float16)zv;
      zbuf[((size_t)b * NROWS + r0 + M) * NLAT + m] = zv;
    }
  }
  __syncthreads();

  // ---- Decoder 1: r = relu(z @ dec_w1 + db1)  (K=16 padded) ----
  {
    v16h a = loadA(&Zh[0][0], 32, 0, half, m);
    v8f acc[2];
#pragma unroll
    for (int t = 0; t < 2; ++t) {
      v8f c = {};
      acc[t] = wmma16(a, loadB(DW1T, 32, t * 16, 0, half, m), c);
    }
    __syncthreads();  // finish h2 reads before overwriting Ah with r
#pragma unroll
    for (int t = 0; t < 2; ++t)
#pragma unroll
      for (int r = 0; r < 8; ++r) {
        int M = r + half * 8, n = t * 16 + m;
        float v = acc[t][r] + db1[n];
        Ah[M][n] = (_Float16)(v > 0.0f ? v : 0.0f);
      }
  }
  __syncthreads();

  // ---- Decoder 2: recon = relu(r @ dec_w2 + db2) + loss partial ----
  {
    v16h a = loadA(&Ah[0][0], 64, 0, half, m);
    v8f c = {};
    c = wmma16(a, loadB(DW2T, 32, 0, 0, half, m), c);
    float bias = db2[m];
    float lsum = 0.0f;
#pragma unroll
    for (int r = 0; r < 8; ++r) {
      int M = r + half * 8;
      float v = c[r] + bias;
      v = v > 0.0f ? v : 0.0f;
      float d = v - Xg[M * NCOLS + m];
      lsum += d * d;
    }
#pragma unroll
    for (int off = 16; off; off >>= 1) lsum += __shfl_xor(lsum, off, 32);
    if (lane == 0) loss_partials[blk] = lsum;  // deterministic: no atomics
  }
}

// ---------------------------------------------------------------------------
// Stage 2: per-batch cluster centers, d2, min/max normalize, scores; loss.
// Deterministic: one thread per (class,dim) slot scans rows in fixed order.
// ---------------------------------------------------------------------------
__global__ __launch_bounds__(512) void centers_scores_kernel(
    const float* __restrict__ zbuf, const int* __restrict__ labels,
    float* __restrict__ scores_out, const float* __restrict__ loss_partials,
    float* __restrict__ loss_out) {
  __shared__ float cent[NCLASS * NLAT];
  __shared__ float cnt[NCLASS];
  __shared__ float d2s[NROWS];
  __shared__ float rmin[512];
  __shared__ float rmax[512];

  const int b = blockIdx.x, tid = threadIdx.x;
  const float* zb = zbuf + (size_t)b * NROWS * NLAT;

  if (tid < NCLASS * NLAT) {  // 512 slots exactly
    int k = tid >> 4, d = tid & 15;
    float s = 0.0f;
    for (int r = 0; r < NROWS; ++r)
      if (labels[r] == k) s += zb[r * NLAT + d];
    cent[tid] = s;
  }
  if (tid < NCLASS) {
    float c = 0.0f;
    for (int r = 0; r < NROWS; ++r)
      if (labels[r] == tid) c += 1.0f;
    cnt[tid] = c > 1.0f ? c : 1.0f;
  }
  __syncthreads();
  if (tid < NCLASS * NLAT) cent[tid] /= cnt[tid >> 4];
  __syncthreads();

  float lmn = 3.0e38f, lmx = -3.0e38f;
  for (int r = tid; r < NROWS; r += 512) {
    int lab = labels[r];
    float s = 0.0f;
#pragma unroll
    for (int d = 0; d < NLAT; ++d) {
      float df = zb[r * NLAT + d] - cent[lab * NLAT + d];
      s += df * df;
    }
    s *= (1.0f / NLAT);
    d2s[r] = s;
    lmn = fminf(lmn, s);
    lmx = fmaxf(lmx, s);
  }
  rmin[tid] = lmn;
  rmax[tid] = lmx;
  __syncthreads();
  for (int off = 256; off; off >>= 1) {
    if (tid < off) {
      rmin[tid] = fminf(rmin[tid], rmin[tid + off]);
      rmax[tid] = fmaxf(rmax[tid], rmax[tid + off]);
    }
    __syncthreads();
  }
  const float mn = rmin[0], mx = rmax[0];
  const float inv = 1.0f / (mx - mn);
  for (int r = tid; r < NROWS; r += 512)
    scores_out[(size_t)b * NROWS + r] = (d2s[r] - mn) * inv + (float)labels[r];

  if (b == 0 && tid == 0) {  // deterministic serial loss reduce (768 values)
    float acc = 0.0f;
    for (int i = 0; i < NBLOCKS; ++i) acc += loss_partials[i];
    *loss_out = acc * (1.0f / ((float)NB * NROWS * NCOLS));
  }
}

// ---------------------------------------------------------------------------
// Stage 3: per-batch bitonic sort (2048-wide, index tiebreak) + O(n) PAV
// isotonic regression == reference's cummin/cummax diag soft-rank.
// ---------------------------------------------------------------------------
__global__ __launch_bounds__(1024) void sort_rank_kernel(
    const float* __restrict__ scores_in, const int* __restrict__ blocksize,
    float* __restrict__ ranks_out, float* __restrict__ rankidx_out) {
  __shared__ float key[SORTN];
  __shared__ int   id[SORTN];
  __shared__ float yv[NROWS];
  __shared__ float gv[NROWS];
  __shared__ float bsum[NROWS];
  __shared__ int   bcnt[NROWS];

  const int b = blockIdx.x;
  const unsigned tid = threadIdx.x;
  const float* sc = scores_in + (size_t)b * NROWS;

  for (unsigned i = tid; i < SORTN; i += 1024) {
    if (i < NROWS) { key[i] = sc[i]; id[i] = (int)i; }
    else           { key[i] = 3.0e38f; id[i] = 0x7fffffff; }
  }
  __syncthreads();

  // Bitonic sort ascending by (key, idx).
  for (unsigned size = 2; size <= SORTN; size <<= 1) {
    for (unsigned stride = size >> 1; stride > 0; stride >>= 1) {
      unsigned lo = ((tid & ~(stride - 1)) << 1) | (tid & (stride - 1));
      unsigned hi = lo + stride;
      bool asc = ((lo & size) == 0);
      float ka = key[lo], kb = key[hi];
      int ia = id[lo], ib = id[hi];
      bool gt = (ka > kb) || (ka == kb && ia > ib);
      if (gt == asc) { key[lo] = kb; key[hi] = ka; id[lo] = ib; id[hi] = ia; }
      __syncthreads();
    }
  }

  // rank_idx = ascending rank // BlockSize + 1
  const int bs = blocksize[0];
  for (unsigned s = tid; s < NROWS; s += 1024)
    rankidx_out[(size_t)b * NROWS + id[s]] = (float)((int)s / bs + 1);

  // Soft-rank: descending x = (score - mn)/(mx - mn) * B / eps
  const float smn = key[0], smx = key[NROWS - 1];
  const float scalef = ((float)NB / EPS_REG) / (smx - smn);
  for (unsigned i = tid; i < NROWS; i += 1024) {
    float xs = (key[NROWS - 1 - i] - smn) * scalef;  // descending order
    yv[i] = (float)(NROWS - (int)i) - xs;            // y = w - xs, w = n..1
  }
  __syncthreads();

  // Pool-adjacent-violators: isotonic nondecreasing fit of y (exact, O(n)).
  if (tid == 0) {
    int top = -1;
    for (int i = 0; i < NROWS; ++i) {
      float s = yv[i];
      int c = 1;
      while (top >= 0 && bsum[top] * (float)c >= s * (float)bcnt[top]) {
        s += bsum[top];
        c += bcnt[top];
        --top;
      }
      ++top;
      bsum[top] = s;
      bcnt[top] = c;
    }
    int pos = 0;
    for (int j = 0; j <= top; ++j) {
      float mean = bsum[j] / (float)bcnt[j];
      for (int q = 0; q < bcnt[j]; ++q) gv[pos++] = mean;
    }
  }
  __syncthreads();

  // primal = xs + g, scattered back through the (descending) permutation.
  for (unsigned i = tid; i < NROWS; i += 1024) {
    float xs = (key[NROWS - 1 - i] - smn) * scalef;
    int orig = id[NROWS - 1 - i];
    ranks_out[(size_t)b * NROWS + orig] = xs + gv[i];
  }
}

// ---------------------------------------------------------------------------
extern "C" void kernel_launch(void* const* d_in, const int* in_sizes, int n_in,
                              void* d_out, int out_size, void* d_ws,
                              size_t ws_size, hipStream_t stream) {
  const float* table = (const float*)d_in[0];
  const float* ew1   = (const float*)d_in[1];
  const float* eb1   = (const float*)d_in[2];
  const float* ew2   = (const float*)d_in[3];
  const float* eb2   = (const float*)d_in[4];
  const float* mw    = (const float*)d_in[5];
  const float* mb    = (const float*)d_in[6];
  const float* dw1   = (const float*)d_in[7];
  const float* db1   = (const float*)d_in[8];
  const float* dw2   = (const float*)d_in[9];
  const float* db2   = (const float*)d_in[10];
  const int* labels  = (const int*)d_in[11];
  const int* bs      = (const int*)d_in[12];
  (void)in_sizes; (void)n_in; (void)out_size; (void)ws_size;

  float* out     = (float*)d_out;
  float* ranks   = out;                   // (8,1536,1)
  float* rankidx = out + NB * NROWS;      // (8,1536,1) as float
  float* scores  = out + 2 * NB * NROWS;  // (8,1536,1)
  float* loss    = out + 3 * NB * NROWS;  // scalar

  float* zbuf          = (float*)d_ws;                       // 196608 floats
  float* loss_partials = zbuf + (size_t)NB * NROWS * NLAT;   // 768 floats
  _Float16* blob = (_Float16*)(loss_partials + NBLOCKS);     // 5632 halfs
  // blob byte offset = 789504, 32B-aligned: v16h loads are naturally aligned.

  prep_weights_kernel<<<1, 256, 0, stream>>>(ew1, ew2, mw, dw1, dw2, blob);
  mlp_wmma_kernel<<<NBLOCKS, 32, 0, stream>>>(table, blob, eb1, eb2, mb, db1,
                                              db2, zbuf, loss_partials);
  centers_scores_kernel<<<NB, 512, 0, stream>>>(zbuf, labels, scores,
                                                loss_partials, loss);
  sort_rank_kernel<<<NB, 1024, 0, stream>>>(scores, bs, ranks, rankidx);
}
